// RectangularityLoss_36636071035624
// MI455X (gfx1250) — compile-verified, matrix operand order checked by reading, not run on hardware
//
#include <hip/hip_runtime.h>

typedef __attribute__((ext_vector_type(16))) _Float16 v16h;
typedef __attribute__((ext_vector_type(8)))  float    v8f;

#define BATCH 128
#define H 512
#define W 512
#define ROWS_PER_BLOCK 16
#define BANDS (H / ROWS_PER_BLOCK)      // 32
#define LDS_STRIDE 516                  // floats: bank-conflict-free, 16B-aligned rows
#define WS_STRIDE 64                    // ints per batch: [0..15] rowmask, [16..31] colmask, [32] count

__device__ __forceinline__ void wait_asynccnt0() {
#if __has_builtin(__builtin_amdgcn_s_wait_asynccnt)
    __builtin_amdgcn_s_wait_asynccnt(0);
#else
    asm volatile("s_wait_asynccnt 0x0" ::: "memory");
#endif
}

__global__ __launch_bounds__(256) void rect_init_ws(int* ws) {
    int i = blockIdx.x * 256 + threadIdx.x;
    if (i < BATCH * WS_STRIDE) ws[i] = 0;
}

__global__ __launch_bounds__(256) void rect_scan(const float* __restrict__ pred,
                                                 int* __restrict__ ws) {
    __shared__ float tile[ROWS_PER_BLOCK * LDS_STRIDE];

    const int tid  = threadIdx.x;
    const int band = blockIdx.x & (BANDS - 1);
    const int b    = blockIdx.x >> 5;            // BANDS == 32
    const int r0   = band * ROWS_PER_BLOCK;
    const float* src = pred + ((size_t)b * H + (size_t)r0) * W;

    // ---- stage 16x512 f32 band straight into LDS via CDNA5 async-to-LDS DMA ----
    // global_load_async_to_lds_b128: 16B per lane, global -> LDS, no VGPR round-trip,
    // tracked by ASYNCcnt. LDS byte address = low 32 bits of the flat LDS pointer.
    {
        const unsigned lds_base = (unsigned)(size_t)(&tile[0]);
        #pragma unroll
        for (int i = 0; i < 8; ++i) {
            int f4 = i * 256 + tid;              // float4 index in 16x512 band
            int f  = f4 * 4;
            int r  = f >> 9;                     // / 512
            int c  = f & (W - 1);
            unsigned lds_off = lds_base + (unsigned)((r * LDS_STRIDE + c) * 4);
            unsigned long long ga = (unsigned long long)(const void*)(src + f);
            asm volatile("global_load_async_to_lds_b128 %0, %1, off"
                         :: "v"(lds_off), "v"(ga)
                         : "memory");
        }
        wait_asynccnt0();
    }
    __syncthreads();

    int* wsb = ws + b * WS_STRIDE;

    // ---- row path (wave 0): row sums of the 16x512 band via WMMA ----
    // A tile = 16x32 binarized f16, B = ones(32x16)  =>  D[m][n] = rowsum[m].
    // B all-ones makes K-order irrelevant: lane L only needs 16 values of row L%16.
    if (tid < 32) {
        const int lane = tid;
        const int r    = lane & 15;
        const int half = lane >> 4;              // which 16-col sub-chunk this lane feeds
        v8f acc = {};
        v16h ones;
        #pragma unroll
        for (int j = 0; j < 16; ++j) ones[j] = (_Float16)1.0f;

        #pragma unroll
        for (int c0 = 0; c0 < W; c0 += 32) {
            const float* p = &tile[r * LDS_STRIDE + c0 + half * 16];
            v16h a;
            #pragma unroll
            for (int j = 0; j < 16; ++j)
                a[j] = (p[j] > 0.5f) ? (_Float16)1.0f : (_Float16)0.0f;
            acc = __builtin_amdgcn_wmma_f32_16x16x32_f16(
                false, a, false, ones, (short)0, acc, false, false);
        }
        // C/D layout: VGPR g, lanes 0-15 -> M=g ; lanes 16-31 -> M=g+8.
        // Lane 0 holds rowsums of rows 0..7 (N=0), lane 16 rows 8..15.
        if ((lane & 15) == 0) {
            unsigned bits = 0;
            int cnt = 0;
            #pragma unroll
            for (int g = 0; g < 8; ++g) {
                float s = acc[g];                // exact small integer in f32
                cnt += (int)(s + 0.5f);
                if (s > 0.5f) bits |= 1u << (g + 8 * half);
            }
            atomicOr((unsigned*)&wsb[r0 >> 5], bits << (r0 & 31));
            atomicAdd(&wsb[32], cnt);
        }
    }

    // ---- column path (all 8 waves): col-any via wave32 ballot ----
    {
        bool a0 = false, a1 = false;
        #pragma unroll
        for (int h = 0; h < ROWS_PER_BLOCK; ++h) {
            a0 |= (tile[h * LDS_STRIDE + tid]       > 0.5f);
            a1 |= (tile[h * LDS_STRIDE + tid + 256] > 0.5f);
        }
        unsigned m0 = __builtin_amdgcn_ballot_w32(a0);   // cols wave*32 .. +31
        unsigned m1 = __builtin_amdgcn_ballot_w32(a1);   // cols 256 + wave*32 .. +31
        int wave = tid >> 5;
        if ((tid & 31) == 0) {
            if (m0) atomicOr((unsigned*)&wsb[16 + wave], m0);
            if (m1) atomicOr((unsigned*)&wsb[24 + wave], m1);
        }
    }
}

__global__ __launch_bounds__(BATCH) void rect_finalize(const int* __restrict__ ws,
                                                       float* __restrict__ out) {
    __shared__ int partial[BATCH];
    const int b = threadIdx.x;
    const int* wsb = ws + b * WS_STRIDE;

    // rows: first/last set bit over 512-bit mask; argmax(all-false)==0 semantics
    int top = 0, bottom = H - 1;
    {
        int first = -1, last = -1;
        for (int i = 0; i < 16; ++i) {
            unsigned w = (unsigned)wsb[i];
            if (w) {
                if (first < 0) first = i * 32 + __builtin_ctz(w);
                last = i * 32 + 31 - __builtin_clz(w);
            }
        }
        if (first >= 0) { top = first; bottom = last; }
    }
    int left = 0, right = W - 1;
    {
        int first = -1, last = -1;
        for (int i = 0; i < 16; ++i) {
            unsigned w = (unsigned)wsb[16 + i];
            if (w) {
                if (first < 0) first = i * 32 + __builtin_ctz(w);
                last = i * 32 + 31 - __builtin_clz(w);
            }
        }
        if (first >= 0) { left = first; right = last; }
    }

    int area = (bottom - top + 1) * (right - left + 1);
    partial[b] = area - wsb[32];                 // |bbox - bin| summed == area - count
    __syncthreads();

    if (b == 0) {
        long long total = 0;
        for (int i = 0; i < BATCH; ++i) total += partial[i];
        out[0] = (float)((double)total / (double)BATCH);   // WEIGHT == 1.0
    }
}

extern "C" void kernel_launch(void* const* d_in, const int* in_sizes, int n_in,
                              void* d_out, int out_size, void* d_ws, size_t ws_size,
                              hipStream_t stream) {
    (void)in_sizes; (void)n_in; (void)out_size; (void)ws_size;
    const float* pred = (const float*)d_in[0];
    float* out = (float*)d_out;
    int* ws = (int*)d_ws;

    rect_init_ws<<<(BATCH * WS_STRIDE + 255) / 256, 256, 0, stream>>>(ws);
    rect_scan<<<BATCH * BANDS, 256, 0, stream>>>(pred, ws);
    rect_finalize<<<1, BATCH, 0, stream>>>(ws, out);
}
